// DecLayer_25417616458199
// MI455X (gfx1250) — compile-verified, matrix operand order checked by reading, not run on hardware
//
#include <hip/hip_runtime.h>
#include <hip/hip_bf16.h>

// ---------------- problem dims ----------------
#define V_   32000
#define E_   256
#define D_   512
#define B_   32
#define T_   128
#define S_   128
#define PAD_ 3
#define G4D_ (4*D_)   // 2048

typedef __bf16 bf16_t;
typedef __bf16 v16bf __attribute__((ext_vector_type(16)));
typedef __bf16 v8bf  __attribute__((ext_vector_type(8)));
typedef float  v8f   __attribute__((ext_vector_type(8)));
typedef unsigned int v4u __attribute__((ext_vector_type(4)));
typedef int    v8i   __attribute__((ext_vector_type(8)));
typedef int    v4i   __attribute__((ext_vector_type(4)));

// round-to-nearest-even f32 -> bf16 via bit ops
__device__ __forceinline__ bf16_t to_bf16(float f) {
    union { float f; unsigned u; } v; v.f = f;
    unsigned r = v.u + 0x7FFFu + ((v.u >> 16) & 1u);
    unsigned short h = (unsigned short)(r >> 16);
    bf16_t o; __builtin_memcpy(&o, &h, 2);
    return o;
}

__device__ __forceinline__ float sigmoidf_(float x) {
    return 1.0f / (1.0f + __expf(-x));
}

// Load one WMMA 16x16x32 bf16 fragment row/col for this lane.
// Lane layout (wave32): half = lane>>4; holds K in {8h..8h+7} U {16+8h..16+8h+7}
__device__ __forceinline__ v16bf load_frag(const bf16_t* p, int half) {
    const bf16_t* q = p + 8 * half;
    v8bf lo = *(const v8bf*)(q);
    v8bf hi = *(const v8bf*)(q + 16);
    v16bf r;
#pragma unroll
    for (int i = 0; i < 8; ++i) { r[i] = lo[i]; r[8 + i] = hi[i]; }
    return r;
}

__device__ __forceinline__ v8f wmma_bf16(v16bf a, v16bf b, v8f c) {
    return __builtin_amdgcn_wmma_f32_16x16x32_bf16(false, a, false, b, (short)0, c, false, false);
}

// ---------------- prep kernels (once per launch) ----------------
__global__ __launch_bounds__(256) void k_prep_wout(const float* __restrict__ W, bf16_t* __restrict__ O) {
    size_t i = (size_t)blockIdx.x * 256 + threadIdx.x;
    if (i < (size_t)V_ * 2 * D_) O[i] = to_bf16(W[i]);
}

__global__ __launch_bounds__(256) void k_prep_emb(const float* __restrict__ E, bf16_t* __restrict__ O) {
    size_t i = (size_t)blockIdx.x * 256 + threadIdx.x;
    if (i < (size_t)V_ * E_) {
        int row = (int)(i / E_);
        O[i] = (row == PAD_) ? to_bf16(0.0f) : to_bf16(E[i]);
    }
}

__global__ __launch_bounds__(256) void k_prep_wcat0(const float* __restrict__ Wih, const float* __restrict__ Whh,
                                                    bf16_t* __restrict__ O) {
    size_t i = (size_t)blockIdx.x * 256 + threadIdx.x;
    if (i < (size_t)G4D_ * (E_ + D_)) {
        int n = (int)(i / (E_ + D_)), k = (int)(i % (E_ + D_));
        float v = (k < E_) ? Wih[(size_t)n * E_ + k] : Whh[(size_t)n * D_ + (k - E_)];
        O[i] = to_bf16(v);
    }
}

__global__ __launch_bounds__(256) void k_prep_wcat1(const float* __restrict__ Wih, const float* __restrict__ Whh,
                                                    bf16_t* __restrict__ O) {
    size_t i = (size_t)blockIdx.x * 256 + threadIdx.x;
    if (i < (size_t)G4D_ * (2 * D_)) {
        int n = (int)(i / (2 * D_)), k = (int)(i % (2 * D_));
        float v = (k < D_) ? Wih[(size_t)n * D_ + k] : Whh[(size_t)n * D_ + (k - D_)];
        O[i] = to_bf16(v);
    }
}

__global__ __launch_bounds__(256) void k_prep_misc(const float* __restrict__ s0, const float* __restrict__ c0,
                                                   const float* __restrict__ bih0, const float* __restrict__ bhh0,
                                                   const float* __restrict__ bih1, const float* __restrict__ bhh1,
                                                   bf16_t* s0bf0, bf16_t* s1bf0, float* c0f0, float* c1f0,
                                                   float* b0, float* b1) {
    int i = blockIdx.x * 256 + threadIdx.x;
    if (i < B_ * D_) {
        s0bf0[i] = to_bf16(s0[i]);
        s1bf0[i] = to_bf16(s0[B_ * D_ + i]);
        c0f0[i]  = c0[i];
        c1f0[i]  = c0[B_ * D_ + i];
    }
    if (i < G4D_) {
        b0[i] = bih0[i] + bhh0[i];
        b1[i] = bih1[i] + bhh1[i];
    }
}

// ---------------- LSTM layer 0 ----------------
__global__ __launch_bounds__(128)
void k_lstm0(const int* __restrict__ ids, int t,
             const bf16_t* __restrict__ embb, const bf16_t* __restrict__ Wc,
             const float* __restrict__ bias,
             const bf16_t* __restrict__ s_prev, const float* __restrict__ c_prev,
             bf16_t* __restrict__ s_out, float* __restrict__ c_out) {
    __shared__ float gl[4][B_][32];
    const int tid  = threadIdx.x;
    const int g    = tid >> 5;
    const int lane = tid & 31;
    const int half = lane >> 4, ml = lane & 15;
    const int j0   = blockIdx.x * 32;
    const int ncol = g * D_ + j0;

    const int id0 = ids[ml * T_ + t];
    const int id1 = ids[(16 + ml) * T_ + t];

    v8f acc00 = {}, acc01 = {}, acc10 = {}, acc11 = {};
#pragma unroll 4
    for (int kt = 0; kt < (E_ + D_) / 32; ++kt) {
        const int k0 = kt * 32;
        v16bf a0, a1;
        if (k0 < E_) {
            a0 = load_frag(embb + (size_t)id0 * E_ + k0, half);
            a1 = load_frag(embb + (size_t)id1 * E_ + k0, half);
        } else {
            const int ks = k0 - E_;
            a0 = load_frag(s_prev + ml * D_ + ks, half);
            a1 = load_frag(s_prev + (16 + ml) * D_ + ks, half);
        }
        v16bf w0 = load_frag(Wc + (size_t)(ncol + ml) * (E_ + D_) + k0, half);
        v16bf w1 = load_frag(Wc + (size_t)(ncol + 16 + ml) * (E_ + D_) + k0, half);
        acc00 = wmma_bf16(a0, w0, acc00);
        acc01 = wmma_bf16(a1, w0, acc01);
        acc10 = wmma_bf16(a0, w1, acc10);
        acc11 = wmma_bf16(a1, w1, acc11);
    }
    {
        float bv0 = bias[ncol + ml];
        float bv1 = bias[ncol + 16 + ml];
#pragma unroll
        for (int r = 0; r < 8; ++r) {
            int b = half * 8 + r;
            gl[g][b][ml]           = acc00[r] + bv0;
            gl[g][16 + b][ml]      = acc01[r] + bv0;
            gl[g][b][16 + ml]      = acc10[r] + bv1;
            gl[g][16 + b][16 + ml] = acc11[r] + bv1;
        }
    }
    __syncthreads();
    for (int e = tid; e < B_ * 32; e += 128) {
        int b = e >> 5, jj = e & 31;
        int j = j0 + jj;
        float iv = sigmoidf_(gl[0][b][jj]);
        float fv = sigmoidf_(gl[1][b][jj]);
        float gv = tanhf(gl[2][b][jj]);
        float ov = sigmoidf_(gl[3][b][jj]);
        float c2 = fv * c_prev[b * D_ + j] + iv * gv;
        float s2 = ov * tanhf(c2);
        c_out[b * D_ + j] = c2;
        s_out[b * D_ + j] = to_bf16(s2);
    }
}

// ---------------- LSTM layer 1 ----------------
__global__ __launch_bounds__(128)
void k_lstm1(const bf16_t* __restrict__ s_low, const bf16_t* __restrict__ s_prev,
             const bf16_t* __restrict__ Wc, const float* __restrict__ bias,
             const float* __restrict__ c_prev,
             bf16_t* __restrict__ s_out, float* __restrict__ c_out,
             float* __restrict__ s_out_f, bf16_t* __restrict__ x_bf) {
    __shared__ float gl[4][B_][32];
    const int tid  = threadIdx.x;
    const int g    = tid >> 5;
    const int lane = tid & 31;
    const int half = lane >> 4, ml = lane & 15;
    const int j0   = blockIdx.x * 32;
    const int ncol = g * D_ + j0;

    v8f acc00 = {}, acc01 = {}, acc10 = {}, acc11 = {};
#pragma unroll 4
    for (int kt = 0; kt < (2 * D_) / 32; ++kt) {
        const int k0 = kt * 32;
        v16bf a0, a1;
        if (k0 < D_) {
            a0 = load_frag(s_low + ml * D_ + k0, half);
            a1 = load_frag(s_low + (16 + ml) * D_ + k0, half);
        } else {
            const int ks = k0 - D_;
            a0 = load_frag(s_prev + ml * D_ + ks, half);
            a1 = load_frag(s_prev + (16 + ml) * D_ + ks, half);
        }
        v16bf w0 = load_frag(Wc + (size_t)(ncol + ml) * (2 * D_) + k0, half);
        v16bf w1 = load_frag(Wc + (size_t)(ncol + 16 + ml) * (2 * D_) + k0, half);
        acc00 = wmma_bf16(a0, w0, acc00);
        acc01 = wmma_bf16(a1, w0, acc01);
        acc10 = wmma_bf16(a0, w1, acc10);
        acc11 = wmma_bf16(a1, w1, acc11);
    }
    {
        float bv0 = bias[ncol + ml];
        float bv1 = bias[ncol + 16 + ml];
#pragma unroll
        for (int r = 0; r < 8; ++r) {
            int b = half * 8 + r;
            gl[g][b][ml]           = acc00[r] + bv0;
            gl[g][16 + b][ml]      = acc01[r] + bv0;
            gl[g][b][16 + ml]      = acc10[r] + bv1;
            gl[g][16 + b][16 + ml] = acc11[r] + bv1;
        }
    }
    __syncthreads();
    for (int e = tid; e < B_ * 32; e += 128) {
        int b = e >> 5, jj = e & 31;
        int j = j0 + jj;
        float iv = sigmoidf_(gl[0][b][jj]);
        float fv = sigmoidf_(gl[1][b][jj]);
        float gv = tanhf(gl[2][b][jj]);
        float ov = sigmoidf_(gl[3][b][jj]);
        float c2 = fv * c_prev[b * D_ + j] + iv * gv;
        float s2 = ov * tanhf(c2);
        c_out[b * D_ + j]   = c2;
        s_out[b * D_ + j]   = to_bf16(s2);
        s_out_f[b * D_ + j] = s2;
        x_bf[b * (2 * D_) + j] = to_bf16(s2);
    }
}

// ---------------- attention ----------------
__global__ __launch_bounds__(128)
void k_attn(const float* __restrict__ h, const float* __restrict__ s1n, bf16_t* __restrict__ x_bf) {
    __shared__ float sc[S_];
    __shared__ float red[128];
    const int b = blockIdx.x, tid = threadIdx.x;
    const float* hb = h + (size_t)b * S_ * D_;
    const float* sb = s1n + (size_t)b * D_;

    float acc = 0.f;
    const float* hr = hb + (size_t)tid * D_;
    for (int d = 0; d < D_; d += 4) {
        float4 hv = *(const float4*)(hr + d);
        float4 sv = *(const float4*)(sb + d);
        acc += hv.x * sv.x + hv.y * sv.y + hv.z * sv.z + hv.w * sv.w;
    }
    red[tid] = acc; __syncthreads();
    for (int o = 64; o > 0; o >>= 1) { if (tid < o) red[tid] = fmaxf(red[tid], red[tid + o]); __syncthreads(); }
    const float mx = red[0]; __syncthreads();
    const float ex = __expf(acc - mx);
    sc[tid] = ex; red[tid] = ex; __syncthreads();
    for (int o = 64; o > 0; o >>= 1) { if (tid < o) red[tid] += red[tid + o]; __syncthreads(); }
    const float inv = 1.f / red[0];
    __syncthreads();

    const int d0 = tid * 4;
    float z0 = 0, z1 = 0, z2 = 0, z3 = 0;
    for (int s = 0; s < S_; ++s) {
        float a = sc[s] * inv;
        float4 hv = *(const float4*)(hb + (size_t)s * D_ + d0);
        z0 += a * hv.x; z1 += a * hv.y; z2 += a * hv.z; z3 += a * hv.w;
    }
    bf16_t* xo = x_bf + (size_t)b * (2 * D_) + D_ + d0;
    xo[0] = to_bf16(z0); xo[1] = to_bf16(z1); xo[2] = to_bf16(z2); xo[3] = to_bf16(z3);
}

// ---------------- output projection ----------------
// grid: V_/128 = 250 WGs, 128 threads (4 waves; each wave -> 32x32 output tile)
// x (32x1024 bf16, 64KB) staged into LDS via Tensor Data Mover (wave 0), TENSORcnt-waited.
__global__ __launch_bounds__(128)
void k_outproj(const bf16_t* __restrict__ x_bf, const bf16_t* __restrict__ Wb,
               const float* __restrict__ bout, float* __restrict__ Lws, int t) {
    __shared__ bf16_t xs[B_ * 2 * D_];   // 32x1024 bf16 = 64 KB
    const int tid = threadIdx.x;
    const int wave = tid >> 5, lane = tid & 31;

    if (wave == 0) {
        // Tensor DMA descriptor (D#): 2D tile, data_size=2B, tile 1024x32 == tensor, stride 1024
        const unsigned long long ga = (unsigned long long)(const void*)x_bf;
        const unsigned lds_base = (unsigned)(unsigned long long)(void*)&xs[0];
        v4u g0;
        g0[0] = 1u;                                              // count=1, user mode
        g0[1] = lds_base;                                        // lds_addr
        g0[2] = (unsigned)(ga & 0xFFFFFFFFull);                  // global_addr lo
        g0[3] = (unsigned)((ga >> 32) & 0x01FFFFFFull) | (2u << 30); // global_addr hi | type=2
        v8i g1;
        g1[0] = (int)(1u << 16);                                 // workgroup_mask=0, data_size=1 (2B)
        g1[1] = (int)((1024u & 0xFFFFu) << 16);                  // tensor_dim0 lo16 (=1024)
        g1[2] = (int)((1024u >> 16) | (32u << 16));              // tensor_dim0 hi16 | tensor_dim1 lo16 (=32)
        g1[3] = (int)(1024u << 16);                              // tensor_dim1 hi16 | tile_dim0 (=1024)
        g1[4] = 32;                                              // tile_dim1 (=32), tile_dim2=0
        g1[5] = 1024;                                            // tensor_dim0_stride lo32
        g1[6] = 0;                                               // stride0 hi16 | stride1 lo16
        g1[7] = 0;
        v4i g2 = {}, g3 = {};
        v8i g4 = {};
        __builtin_amdgcn_tensor_load_to_lds(g0, g1, g2, g3, g4, 0);
        __builtin_amdgcn_s_wait_tensorcnt(0);
    }
    __syncthreads();

    const int half = lane >> 4, ml = lane & 15;
    const int v0 = blockIdx.x * 128 + wave * 32;
    const bf16_t* wrow0 = Wb + (size_t)(v0 + ml) * (2 * D_);
    const bf16_t* wrow1 = Wb + (size_t)(v0 + 16 + ml) * (2 * D_);
    __builtin_prefetch(wrow0, 0, 1);
    __builtin_prefetch(wrow1, 0, 1);

    v8f acc00 = {}, acc01 = {}, acc10 = {}, acc11 = {};
#pragma unroll 2
    for (int kt = 0; kt < (2 * D_) / 32; ++kt) {
        const int k0 = kt * 32;
        v16bf a0 = load_frag(xs + ml * (2 * D_) + k0, half);
        v16bf a1 = load_frag(xs + (16 + ml) * (2 * D_) + k0, half);
        v16bf b0 = load_frag(wrow0 + k0, half);
        v16bf b1 = load_frag(wrow1 + k0, half);
        acc00 = wmma_bf16(a0, b0, acc00);
        acc01 = wmma_bf16(a1, b0, acc01);
        acc10 = wmma_bf16(a0, b1, acc10);
        acc11 = wmma_bf16(a1, b1, acc11);
    }
    const float bo0 = bout[v0 + ml];
    const float bo1 = bout[v0 + 16 + ml];
    float* o0 = Lws + ((size_t)t * B_) * V_ + v0 + ml;
    float* o1 = o0 + 16;
#pragma unroll
    for (int r = 0; r < 8; ++r) {
        int b = half * 8 + r;
        o0[(size_t)b * V_]        = acc00[r] + bo0;
        o0[(size_t)(16 + b) * V_] = acc01[r] + bo0;
        o1[(size_t)b * V_]        = acc10[r] + bo1;
        o1[(size_t)(16 + b) * V_] = acc11[r] + bo1;
    }
}

// ---------------- final transpose: (T,B,V) -> (B,V,T) ----------------
__global__ __launch_bounds__(256)
void k_transpose(const float* __restrict__ Lws, float* __restrict__ out) {
    __shared__ float tile[32][33];
    const int v0 = blockIdx.x * 32, t0 = blockIdx.y * 32, b = blockIdx.z;
    const int tx = threadIdx.x & 31, ty = threadIdx.x >> 5;
#pragma unroll
    for (int i = 0; i < 32; i += 8) {
        int tt = t0 + ty + i;
        tile[ty + i][tx] = Lws[((size_t)tt * B_ + b) * V_ + v0 + tx];
    }
    __syncthreads();
#pragma unroll
    for (int i = 0; i < 32; i += 8) {
        int vv = v0 + ty + i;
        out[((size_t)b * V_ + vv) * T_ + t0 + tx] = tile[tx][ty + i];
    }
}

// ---------------- host launcher ----------------
extern "C" void kernel_launch(void* const* d_in, const int* in_sizes, int n_in,
                              void* d_out, int out_size, void* d_ws, size_t ws_size,
                              hipStream_t stream) {
    const int*   ids  = (const int*)  d_in[0];
    const float* h    = (const float*)d_in[1];
    const float* s0   = (const float*)d_in[2];
    const float* c0   = (const float*)d_in[3];
    const float* emb  = (const float*)d_in[4];
    const float* Wih0 = (const float*)d_in[5];
    const float* Whh0 = (const float*)d_in[6];
    const float* bih0 = (const float*)d_in[7];
    const float* bhh0 = (const float*)d_in[8];
    const float* Wih1 = (const float*)d_in[9];
    const float* Whh1 = (const float*)d_in[10];
    const float* bih1 = (const float*)d_in[11];
    const float* bhh1 = (const float*)d_in[12];
    const float* Wout = (const float*)d_in[13];
    const float* bout = (const float*)d_in[14];
    float* out = (float*)d_out;

    char* w = (char*)d_ws;
    auto alloc = [&](size_t bytes) -> char* {
        char* p = w; w += (bytes + 255) & ~(size_t)255; return p;
    };
    const size_t SD = (size_t)B_ * D_;
    bf16_t* Wout_bf = (bf16_t*)alloc((size_t)V_ * 2 * D_ * 2);
    bf16_t* Wcat0   = (bf16_t*)alloc((size_t)G4D_ * (E_ + D_) * 2);
    bf16_t* Wcat1   = (bf16_t*)alloc((size_t)G4D_ * (2 * D_) * 2);
    bf16_t* emb_bf  = (bf16_t*)alloc((size_t)V_ * E_ * 2);
    float*  b0      = (float*) alloc(G4D_ * 4);
    float*  b1      = (float*) alloc(G4D_ * 4);
    bf16_t* s0bf    = (bf16_t*)alloc(2 * SD * 2);
    bf16_t* s1bf    = (bf16_t*)alloc(2 * SD * 2);
    float*  c0f     = (float*) alloc(2 * SD * 4);
    float*  c1f     = (float*) alloc(2 * SD * 4);
    float*  s1nf    = (float*) alloc(SD * 4);
    bf16_t* x_bf    = (bf16_t*)alloc((size_t)B_ * 2 * D_ * 2);
    float*  Lws     = (float*) alloc((size_t)T_ * B_ * V_ * 4);

    k_prep_wout <<<(int)(((size_t)V_ * 2 * D_ + 255) / 256), 256, 0, stream>>>(Wout, Wout_bf);
    k_prep_emb  <<<(int)(((size_t)V_ * E_ + 255) / 256),     256, 0, stream>>>(emb, emb_bf);
    k_prep_wcat0<<<(G4D_ * (E_ + D_) + 255) / 256,           256, 0, stream>>>(Wih0, Whh0, Wcat0);
    k_prep_wcat1<<<(G4D_ * (2 * D_) + 255) / 256,            256, 0, stream>>>(Wih1, Whh1, Wcat1);
    k_prep_misc <<<(B_ * D_ + 255) / 256,                    256, 0, stream>>>(
        s0, c0, bih0, bhh0, bih1, bhh1, s0bf, s1bf, c0f, c1f, b0, b1);

    for (int t = 0; t < T_; ++t) {
        const int p = t & 1, q = 1 - p;
        k_lstm0<<<D_ / 32, 128, 0, stream>>>(ids, t, emb_bf, Wcat0, b0,
                                             s0bf + p * SD, c0f + p * SD,
                                             s0bf + q * SD, c0f + q * SD);
        k_lstm1<<<D_ / 32, 128, 0, stream>>>(s0bf + q * SD, s1bf + p * SD, Wcat1, b1,
                                             c1f + p * SD,
                                             s1bf + q * SD, c1f + q * SD, s1nf, x_bf);
        k_attn<<<B_, 128, 0, stream>>>(h, s1nf, x_bf);
        k_outproj<<<V_ / 128, 128, 0, stream>>>(x_bf, Wout_bf, bout, Lws, t);
    }
    k_transpose<<<dim3(V_ / 32, T_ / 32, B_), 256, 0, stream>>>(Lws, out);
}
